// TriangleMultiplicationOutgoing_30391188586656
// MI455X (gfx1250) — compile-verified, matrix operand order checked by reading, not run on hardware
//
#include <hip/hip_runtime.h>
#include <hip/hip_bf16.h>

// ---------------------------------------------------------------------------
// TriangleMultiplicationOutgoing for MI455X (gfx1250, wave32, WMMA bf16)
//   z = LN(pair);  A = sig(g1(z))*l1(z)*m;  B = sig(g2(z))*l2(z)*m
//   vals[i,j,c] = sum_k A[i,k,c]*B[j,k,c];  out = sig(gate(z))*lin(LN(vals))*m
//
// Layouts chosen for coalescing + L2 residency (192MB L2):
//   At, Bt : bf16 [c][i*512+k]  (contiguous-k rows per channel for the GEMM)
//   vals   : bf16 [c][i*512+j]  (lane-contiguous stores in k2, 64MB not 128MB)
//   Gt     : bf16 [row][c]
// ---------------------------------------------------------------------------

typedef __bf16 bf16_t;
typedef __attribute__((ext_vector_type(16))) __bf16 v16bf;
typedef __attribute__((ext_vector_type(8)))  __bf16 v8bf;
typedef __attribute__((ext_vector_type(8)))  float  v8f;

#define N_SEQ 512
#define C_DIM 128
#define NN    (N_SEQ * N_SEQ)        // 262144 positions
#define ZSTRIDE 136                  // LDS tile stride (272B: bank-conflict free)

__device__ __forceinline__ v16bf make_frag(v8bf lo, v8bf hi) {
    v16bf r;
#pragma unroll
    for (int i = 0; i < 8; ++i) { r[i] = lo[i]; r[i + 8] = hi[i]; }
    return r;
}

__device__ __forceinline__ float sigmoidf_(float x) {
    return 1.0f / (1.0f + __expf(-x));
}

// ---------------------------------------------------------------------------
// Kernel 0: convert the six 128x128 fp32 weight matrices to bf16 in workspace.
// Order: g1, l1, g2, l2, gate, out
// ---------------------------------------------------------------------------
__global__ __launch_bounds__(256) void convert_weights_kernel(
    const float* __restrict__ g1, const float* __restrict__ l1,
    const float* __restrict__ g2, const float* __restrict__ l2,
    const float* __restrict__ gw, const float* __restrict__ ow,
    bf16_t* __restrict__ wb)
{
    int t   = blockIdx.x * blockDim.x + threadIdx.x;   // 0 .. 6*16384-1
    int mat = t >> 14;
    int idx = t & 16383;
    float v;
    if      (mat == 0) v = g1[idx];
    else if (mat == 1) v = l1[idx];
    else if (mat == 2) v = g2[idx];
    else if (mat == 3) v = l2[idx];
    else if (mat == 4) v = gw[idx];
    else               v = ow[idx];
    wb[t] = (bf16_t)v;
}

// ---------------------------------------------------------------------------
// Kernel 1: fused LayerNorm + five 128->128 projections (WMMA bf16) + gating.
// One block = 16 rows of the flattened (i,k) grid.  8 waves x 16 output cols.
// A/B written as packed v8bf (each lane holds 8 consecutive rows of one
// channel); G transposed through per-wave LDS to get 16B stores.
// ---------------------------------------------------------------------------
__global__ __launch_bounds__(256) void ln_proj_kernel(
    const float* __restrict__ zin,   // [NN][128]
    const float* __restrict__ mask,  // [NN]
    const float* __restrict__ lnw, const float* __restrict__ lnb,
    const bf16_t* __restrict__ wb,   // 6x[128][128] bf16 (g1,l1,g2,l2,gate,out)
    const float* __restrict__ g1b, const float* __restrict__ l1b,
    const float* __restrict__ g2b, const float* __restrict__ l2b,
    const float* __restrict__ gateb,
    bf16_t* __restrict__ At,         // [128][NN]
    bf16_t* __restrict__ Bt,         // [128][NN]
    bf16_t* __restrict__ Gt)         // [NN][128]
{
    __shared__ __align__(16) bf16_t zt[16 * ZSTRIDE];
    __shared__ __align__(16) bf16_t gbuf[8][16][24];   // per-wave G transpose
    __shared__ float reds[16][16];
    __shared__ float redq[16][16];
    __shared__ float mrow[16];

    const int tid     = threadIdx.x;
    const int rowbase = blockIdx.x * 16;
    const int r   = tid >> 4;        // local row 0..15
    const int g   = tid & 15;        // channel group
    const int ch0 = g * 8;

    // ---- LayerNorm over 128 channels (16 lanes x 8 channels per row) ----
    const float* src = zin + (size_t)(rowbase + r) * C_DIM + ch0;
    float x[8];
#pragma unroll
    for (int i = 0; i < 8; ++i) x[i] = src[i];
    float s = 0.f, q = 0.f;
#pragma unroll
    for (int i = 0; i < 8; ++i) { s += x[i]; q += x[i] * x[i]; }
    reds[r][g] = s;
    redq[r][g] = q;
    if (tid < 16) mrow[tid] = mask[rowbase + tid];
    __syncthreads();

    float ts = 0.f, tq = 0.f;
#pragma unroll
    for (int i = 0; i < 16; ++i) { ts += reds[r][i]; tq += redq[r][i]; }
    float mu   = ts * (1.0f / 128.0f);
    float var  = tq * (1.0f / 128.0f) - mu * mu;
    float rstd = rsqrtf(var + 1e-5f);
#pragma unroll
    for (int i = 0; i < 8; ++i) {
        float y = (x[i] - mu) * rstd * lnw[ch0 + i] + lnb[ch0 + i];
        zt[r * ZSTRIDE + ch0 + i] = (bf16_t)y;
    }
    __syncthreads();

    // ---- WMMA: 5 projections, wave 'wid' owns output cols [wid*16, +16) ----
    const int wid   = tid >> 5;
    const int lane  = tid & 31;
    const int hiH   = lane >> 4;     // half-wave select (K sub-range)
    const int lnl   = lane & 15;     // M / N index within fragment
    const int nbase = wid * 16;

    v16bf af[4];
#pragma unroll
    for (int kb4 = 0; kb4 < 4; ++kb4) {
        int k0 = kb4 * 32 + hiH * 8;
        af[kb4] = make_frag(*(const v8bf*)&zt[lnl * ZSTRIDE + k0],
                            *(const v8bf*)&zt[lnl * ZSTRIDE + k0 + 16]);
    }

    v8f acc[5];
#pragma unroll
    for (int p = 0; p < 5; ++p) acc[p] = (v8f){};

#pragma unroll
    for (int p = 0; p < 5; ++p) {
        const bf16_t* W = wb + (size_t)p * C_DIM * C_DIM
                             + (size_t)(nbase + lnl) * C_DIM;
#pragma unroll
        for (int kb4 = 0; kb4 < 4; ++kb4) {
            int k0 = kb4 * 32 + hiH * 8;
            v16bf bfrag = make_frag(*(const v8bf*)&W[k0],
                                    *(const v8bf*)&W[k0 + 16]);
            acc[p] = __builtin_amdgcn_wmma_f32_16x16x32_bf16(
                false, af[kb4], false, bfrag, (short)0, acc[p], false, false);
        }
    }

    // ---- fused sigmoid-gating epilogue ----
    // Lane (lnl,hiH) holds v=0..7 => rows rowbase + hiH*8 + v (contiguous!),
    // channel c = nbase+lnl fixed: A/B pack into one 16B store per lane.
    const int c   = nbase + lnl;
    const float bg1 = g1b[c], bl1 = l1b[c];
    const float bg2 = g2b[c], bl2 = l2b[c];
    const float bga = gateb[c];
    v8bf av, bv, gv;
#pragma unroll
    for (int v = 0; v < 8; ++v) {
        int   m  = v + hiH * 8;
        float mk = mrow[m];
        av[v] = (bf16_t)(sigmoidf_(acc[0][v] + bg1) * (acc[1][v] + bl1) * mk);
        bv[v] = (bf16_t)(sigmoidf_(acc[2][v] + bg2) * (acc[3][v] + bl2) * mk);
        gv[v] = (bf16_t)(sigmoidf_(acc[4][v] + bga));
    }
    const size_t ro = (size_t)c * NN + rowbase + hiH * 8;
    *(v8bf*)(At + ro) = av;
    *(v8bf*)(Bt + ro) = bv;

    // G: transpose [c][row] -> [row][c] through per-wave LDS for 16B stores.
    *(v8bf*)&gbuf[wid][lnl][hiH * 8] = gv;
    __syncthreads();
    v8bf go;
#pragma unroll
    for (int cc = 0; cc < 8; ++cc) go[cc] = gbuf[wid][hiH * 8 + cc][lnl];
    *(v8bf*)(Gt + (size_t)(rowbase + lnl) * C_DIM + nbase + hiH * 8) = go;
}

// ---------------------------------------------------------------------------
// Kernel 2: per-channel triangle contraction  vals[c] = A_c . B_c^T
// 128 channels x 512x512x512 GEMM.  Block = 128x128 tile, 8 waves as 2x4,
// wave = 64x32 (4x2 accumulators => 8 WMMA per 6KB of fragments, ~21 flop/B).
// ---------------------------------------------------------------------------
__global__ __launch_bounds__(256) void triangle_gemm_kernel(
    const bf16_t* __restrict__ At,   // [128][512*512] row-major per channel
    const bf16_t* __restrict__ Bt,   // [128][512*512]
    bf16_t* __restrict__ vals)       // [128][512*512]  ([c][i*512+j])
{
    const int ch = blockIdx.z;
    const int jb = blockIdx.x * 128;
    const int ib = blockIdx.y * 128;
    const bf16_t* Ac = At + (size_t)ch * NN;
    const bf16_t* Bc = Bt + (size_t)ch * NN;

    const int tid  = threadIdx.x;
    const int wid  = tid >> 5;
    const int lane = tid & 31;
    const int hiH  = lane >> 4;
    const int lnl  = lane & 15;
    const int wi   = wid >> 2;       // 0..1 -> 64-row slice
    const int wj   = wid & 3;        // 0..3 -> 32-col slice
    const int m0   = ib + wi * 64;
    const int n0   = jb + wj * 32;

    const bf16_t* arow[4];
    const bf16_t* brow[2];
#pragma unroll
    for (int mt = 0; mt < 4; ++mt)
        arow[mt] = Ac + (size_t)(m0 + mt * 16 + lnl) * N_SEQ + hiH * 8;
#pragma unroll
    for (int nt = 0; nt < 2; ++nt)
        brow[nt] = Bc + (size_t)(n0 + nt * 16 + lnl) * N_SEQ + hiH * 8;

    v8f acc[4][2];
#pragma unroll
    for (int mt = 0; mt < 4; ++mt)
#pragma unroll
        for (int nt = 0; nt < 2; ++nt) acc[mt][nt] = (v8f){};

#pragma unroll 2
    for (int kb = 0; kb < N_SEQ; kb += 32) {
        v16bf afr[4], bfr[2];
#pragma unroll
        for (int mt = 0; mt < 4; ++mt)
            afr[mt] = make_frag(*(const v8bf*)(arow[mt] + kb),
                                *(const v8bf*)(arow[mt] + kb + 16));
#pragma unroll
        for (int nt = 0; nt < 2; ++nt)
            bfr[nt] = make_frag(*(const v8bf*)(brow[nt] + kb),
                                *(const v8bf*)(brow[nt] + kb + 16));
        if (kb + 64 < N_SEQ) {
            __builtin_prefetch(arow[0] + kb + 64, 0, 1);   // global_prefetch_b8
            __builtin_prefetch(brow[0] + kb + 64, 0, 1);
        }
#pragma unroll
        for (int mt = 0; mt < 4; ++mt)
#pragma unroll
            for (int nt = 0; nt < 2; ++nt)
                acc[mt][nt] = __builtin_amdgcn_wmma_f32_16x16x32_bf16(
                    false, afr[mt], false, bfr[nt], (short)0, acc[mt][nt],
                    false, false);
    }

    // Store: [c][i*512+j] -> lanes sweep j (contiguous bf16 across half-wave).
    bf16_t* vc = vals + (size_t)ch * NN;
#pragma unroll
    for (int mt = 0; mt < 4; ++mt) {
#pragma unroll
        for (int nt = 0; nt < 2; ++nt) {
            int j = n0 + nt * 16 + lnl;
#pragma unroll
            for (int v = 0; v < 8; ++v) {
                int i = m0 + mt * 16 + v + 8 * hiH;
                vc[(size_t)i * N_SEQ + j] = (bf16_t)acc[mt][nt][v];
            }
        }
    }
}

// ---------------------------------------------------------------------------
// Kernel 3: LayerNorm(vals) + output projection (WMMA) + gate*mask epilogue.
// Block = 32 consecutive rows (fixed i, j contiguous): the channel-major
// strided loads are coalesced across the 32 row-lanes (full 64B segments).
// ---------------------------------------------------------------------------
__global__ __launch_bounds__(256) void ln_out_kernel(
    const bf16_t* __restrict__ vals,  // [128][NN]
    const bf16_t* __restrict__ Gt,    // [NN][128]
    const float* __restrict__ mask,   // [NN]
    const float* __restrict__ lnw, const float* __restrict__ lnb,
    const bf16_t* __restrict__ Wout,  // [128][128] bf16
    const float* __restrict__ outb,
    float* __restrict__ out)          // [NN][128]
{
    __shared__ __align__(16) bf16_t zt[32 * ZSTRIDE];
    __shared__ float reds[32][9];     // pad 9: conflict-free strided writes
    __shared__ float redq[32][9];
    __shared__ float mrow[32];

    const int tid     = threadIdx.x;
    const int rowbase = blockIdx.x * 32;
    const int rr  = tid & 31;         // local row
    const int cg  = tid >> 5;         // channel group 0..7
    const int c0  = cg * 16;

    // ---- load 16 channels for this row (coalesced: lanes sweep rows) ----
    float x[16];
#pragma unroll
    for (int ci = 0; ci < 16; ++ci)
        x[ci] = (float)vals[(size_t)(c0 + ci) * NN + rowbase + rr];
    float s = 0.f, q = 0.f;
#pragma unroll
    for (int ci = 0; ci < 16; ++ci) { s += x[ci]; q += x[ci] * x[ci]; }
    reds[rr][cg] = s;
    redq[rr][cg] = q;
    if (tid < 32) mrow[tid] = mask[rowbase + tid];
    __syncthreads();

    float ts = 0.f, tq = 0.f;
#pragma unroll
    for (int i = 0; i < 8; ++i) { ts += reds[rr][i]; tq += redq[rr][i]; }
    float mu   = ts * (1.0f / 128.0f);
    float var  = tq * (1.0f / 128.0f) - mu * mu;
    float rstd = rsqrtf(var + 1e-5f);

    v8bf ylo, yhi;
#pragma unroll
    for (int ci = 0; ci < 8; ++ci)
        ylo[ci] = (bf16_t)((x[ci] - mu) * rstd * lnw[c0 + ci] + lnb[c0 + ci]);
#pragma unroll
    for (int ci = 0; ci < 8; ++ci)
        yhi[ci] = (bf16_t)((x[ci + 8] - mu) * rstd * lnw[c0 + 8 + ci] + lnb[c0 + 8 + ci]);
    *(v8bf*)&zt[rr * ZSTRIDE + c0]     = ylo;
    *(v8bf*)&zt[rr * ZSTRIDE + c0 + 8] = yhi;
    __syncthreads();

    // ---- WMMA: wave w owns cols [w*16,+16) for both 16-row tiles ----
    const int wid   = tid >> 5;
    const int lane  = tid & 31;
    const int hiH   = lane >> 4;
    const int lnl   = lane & 15;
    const int nbase = wid * 16;

    // B fragments (weights) loaded once, reused for both row tiles.
    v16bf bfr[4];
    const bf16_t* W = Wout + (size_t)(nbase + lnl) * C_DIM;
#pragma unroll
    for (int kb4 = 0; kb4 < 4; ++kb4) {
        int k0 = kb4 * 32 + hiH * 8;
        bfr[kb4] = make_frag(*(const v8bf*)&W[k0], *(const v8bf*)&W[k0 + 16]);
    }

    v8f acc[2];
#pragma unroll
    for (int rt = 0; rt < 2; ++rt) {
        acc[rt] = (v8f){};
#pragma unroll
        for (int kb4 = 0; kb4 < 4; ++kb4) {
            int k0 = kb4 * 32 + hiH * 8;
            v16bf afrag = make_frag(
                *(const v8bf*)&zt[(rt * 16 + lnl) * ZSTRIDE + k0],
                *(const v8bf*)&zt[(rt * 16 + lnl) * ZSTRIDE + k0 + 16]);
            acc[rt] = __builtin_amdgcn_wmma_f32_16x16x32_bf16(
                false, afrag, false, bfr[kb4], (short)0, acc[rt], false, false);
        }
    }

    const int c  = nbase + lnl;
    const float bo = outb[c];
#pragma unroll
    for (int rt = 0; rt < 2; ++rt) {
#pragma unroll
        for (int v = 0; v < 8; ++v) {
            int   m   = rt * 16 + v + hiH * 8;
            int   row = rowbase + m;
            float gvv = (float)Gt[(size_t)row * C_DIM + c];
            out[(size_t)row * C_DIM + c] = gvv * (acc[rt][v] + bo) * mrow[m];
        }
    }
}

// ---------------------------------------------------------------------------
// Host-side launch
// ---------------------------------------------------------------------------
extern "C" void kernel_launch(void* const* d_in, const int* in_sizes, int n_in,
                              void* d_out, int out_size, void* d_ws, size_t ws_size,
                              hipStream_t stream) {
    const float* z         = (const float*)d_in[0];
    const float* mask      = (const float*)d_in[1];
    const float* ln_pair_w = (const float*)d_in[2];
    const float* ln_pair_b = (const float*)d_in[3];
    const float* ln_out_w  = (const float*)d_in[4];
    const float* ln_out_b  = (const float*)d_in[5];
    const float* g1_w      = (const float*)d_in[6];
    const float* g1_b      = (const float*)d_in[7];
    const float* g2_w      = (const float*)d_in[8];
    const float* g2_b      = (const float*)d_in[9];
    const float* l1_w      = (const float*)d_in[10];
    const float* l1_b      = (const float*)d_in[11];
    const float* l2_w      = (const float*)d_in[12];
    const float* l2_b      = (const float*)d_in[13];
    const float* gate_w    = (const float*)d_in[14];
    const float* gate_b    = (const float*)d_in[15];
    const float* out_w     = (const float*)d_in[16];
    const float* out_b     = (const float*)d_in[17];

    char* ws = (char*)d_ws;
    bf16_t* WB = (bf16_t*)ws;                       // 6 * 128*128 bf16 (192 KB)
    size_t off = 256 * 1024;
    bf16_t* At   = (bf16_t*)(ws + off); off += (size_t)C_DIM * NN * sizeof(bf16_t); // 64 MB
    bf16_t* Bt   = (bf16_t*)(ws + off); off += (size_t)C_DIM * NN * sizeof(bf16_t); // 64 MB
    bf16_t* Gt   = (bf16_t*)(ws + off); off += (size_t)NN * C_DIM * sizeof(bf16_t); // 64 MB
    bf16_t* vals = (bf16_t*)(ws + off);                                             // 64 MB

    convert_weights_kernel<<<(6 * 16384) / 256, 256, 0, stream>>>(
        g1_w, l1_w, g2_w, l2_w, gate_w, out_w, WB);

    ln_proj_kernel<<<NN / 16, 256, 0, stream>>>(
        z, mask, ln_pair_w, ln_pair_b, WB,
        g1_b, l1_b, g2_b, l2_b, gate_b, At, Bt, Gt);

    dim3 g2d(N_SEQ / 128, N_SEQ / 128, C_DIM);
    triangle_gemm_kernel<<<g2d, 256, 0, stream>>>(At, Bt, vals);

    ln_out_kernel<<<NN / 32, 256, 0, stream>>>(
        vals, Gt, mask, ln_out_w, ln_out_b, WB + 5 * 16384, out_b, (float*)d_out);
}
